// GPTClosedLoop_29910152249527
// MI455X (gfx1250) — compile-verified, matrix operand order checked by reading, not run on hardware
//
#include <hip/hip_runtime.h>
#include <hip/hip_bf16.h>

// ---------------------------------------------------------------------------
// GPT closed-loop controller for MI455X (gfx1250), single persistent workgroup.
// Exact reductions vs reference: (1) only batch 0's GPT is observable (plant
// uses Ubuf[0] for all samples), (2) causal GPT + unchanged past inputs =>
// KV-cache incremental decode. Matvecs use v_wmma_f32_16x16x32_f16 with
// weights pre-swizzled into B-fragment layout (ISA 7.12.2). GEMM input
// vectors are kept in LDS as f16 so each A fragment is two ds_load_b128.
// ---------------------------------------------------------------------------

typedef __attribute__((ext_vector_type(16))) _Float16 v16h;
typedef __attribute__((ext_vector_type(8)))  _Float16 v8h;
typedef __attribute__((ext_vector_type(8)))  float    v8f;

#define TSEQ 64
#define NLAYER 4

// fragment storage (halves) per layer: qkv 24*4*512, o 8*4*512, fc 32*4*512, pr 8*16*512
#define FR_QKV_OFF 0
#define FR_O_OFF   49152
#define FR_FC_OFF  65536
#define FR_PR_OFF  131072
#define FR_LAYER   196608
#define FR_TOTAL_HALVES (4*FR_LAYER)            // 786432 halves
#define KV_BYTE_OFF ((size_t)FR_TOTAL_HALVES*2) // 1572864 bytes
#define KV_K_FLOATS 32768                       // 4 layers * 64 t * 128 d

struct GArgs {
  const float *A, *Bm, *Cm, *Dm, *r;
  const float *w_in, *b_in, *wpe, *lnfg, *lnfb, *whead, *bhead;
  const float *wqkv[4], *bqkv[4], *wo[4], *bo[4], *wfc[4], *bfc[4], *wpr[4], *bpr[4];
  const float *l1g[4], *l1b[4], *l2g[4], *l2b[4];
  _Float16 *frag;   // pre-swizzled f16 weight fragments (workspace)
  float    *kv;     // K cache [4][64][128] then V cache [4][64][128]
  float    *Y;      // output (8,64,1) f32
};

__device__ __forceinline__ float geluf(float x) {
  const float c = 0.7978845608028654f; // sqrt(2/pi)
  float t = tanhf(c * (x + 0.044715f * x * x * x));
  return 0.5f * x * (1.0f + t);
}

// Build an A fragment (16x32 f16) whose row 0 is x[kc*32 .. kc*32+31], from an
// f16 LDS vector: exactly two 16-byte ds loads per lane.
//   lane<16 : a[0..7]=x[kc*32+0..7],  a[8..15]=x[kc*32+16..23]
//   lane>=16: a[0..7]=x[kc*32+8..15], a[8..15]=x[kc*32+24..31]
__device__ __forceinline__ v16h buildA(const _Float16* __restrict__ x, int kc) {
  int l = threadIdx.x & 31;
  int off = kc * 32 + ((l < 16) ? 0 : 8);
  v8h lo = *(const v8h*)(x + off);
  v8h hi = *(const v8h*)(x + off + 16);
  v16h a;
#pragma unroll
  for (int j = 0; j < 8; ++j) { a[j] = lo[j]; a[8 + j] = hi[j]; }
  return a;
}

#define MV_STORE 0   // outf[n]  = v      (f32)
#define MV_ADD   1   // outf[n] += v      (f32 residual)
#define MV_GELU  2   // outh[n]  = gelu(v) (f16, feeds next GEMM)

// out[0:N] = mode( x[1xK] @ W[KxN] + bias ), tiles split across the 8 waves.
__device__ __forceinline__ void mv(const _Float16* __restrict__ mat, int nk,
                                   const float* __restrict__ bias,
                                   const _Float16* __restrict__ xin,
                                   float* __restrict__ outf,
                                   _Float16* __restrict__ outh,
                                   int ntiles, int mode) {
  int wv = threadIdx.x >> 5, lane = threadIdx.x & 31;
  if (nk == 4) {
    v16h a0 = buildA(xin, 0), a1 = buildA(xin, 1), a2 = buildA(xin, 2), a3 = buildA(xin, 3);
    for (int nt = wv; nt < ntiles; nt += 8) {
      float bsel = bias[nt * 16 + (lane & 15)];   // unconditional load, select below
      v8f acc = {};
      acc[0] = (lane < 16) ? bsel : 0.0f;
      const _Float16* fb = mat + ((size_t)nt * 4) * 512 + lane * 16;
      v16h b0 = *(const v16h*)(fb);
      acc = __builtin_amdgcn_wmma_f32_16x16x32_f16(false, a0, false, b0, (short)0, acc, false, false);
      v16h b1 = *(const v16h*)(fb + 512);
      acc = __builtin_amdgcn_wmma_f32_16x16x32_f16(false, a1, false, b1, (short)0, acc, false, false);
      v16h b2 = *(const v16h*)(fb + 1024);
      acc = __builtin_amdgcn_wmma_f32_16x16x32_f16(false, a2, false, b2, (short)0, acc, false, false);
      v16h b3 = *(const v16h*)(fb + 1536);
      acc = __builtin_amdgcn_wmma_f32_16x16x32_f16(false, a3, false, b3, (short)0, acc, false, false);
      if (lane < 16) {
        float v = acc[0]; // D row 0, col = lane
        if (mode == MV_ADD)       outf[nt * 16 + lane] += v;
        else if (mode == MV_GELU) outh[nt * 16 + lane]  = (_Float16)geluf(v);
        else                      outf[nt * 16 + lane]  = v;
      }
    }
  } else { // K = 512 (w_pr)
    for (int nt = wv; nt < ntiles; nt += 8) {
      float bsel = bias[nt * 16 + (lane & 15)];
      v8f acc = {};
      acc[0] = (lane < 16) ? bsel : 0.0f;
      const _Float16* fb = mat + ((size_t)nt * nk) * 512 + lane * 16;
      for (int kc = 0; kc < nk; ++kc) {
        v16h a = buildA(xin, kc);
        v16h b = *(const v16h*)(fb + (size_t)kc * 512);
        acc = __builtin_amdgcn_wmma_f32_16x16x32_f16(false, a, false, b, (short)0, acc, false, false);
      }
      if (lane < 16) {
        float v = acc[0];
        if (mode == MV_ADD)       outf[nt * 16 + lane] += v;
        else if (mode == MV_GELU) outh[nt * 16 + lane]  = (_Float16)geluf(v);
        else                      outf[nt * 16 + lane]  = v;
      }
    }
  }
}

// LayerNorm over 128 elements; executed by wave 0 only (caller barriers).
// Writes f32 (for the head dot product) and f16 (for WMMA A fragments).
__device__ __forceinline__ void ln128(const float* __restrict__ in,
                                      float* __restrict__ outf,
                                      _Float16* __restrict__ outh,
                                      const float* __restrict__ g,
                                      const float* __restrict__ b) {
  if ((threadIdx.x >> 5) != 0) return;
  int lane = threadIdx.x & 31;
  float v0 = in[lane], v1 = in[lane + 32], v2 = in[lane + 64], v3 = in[lane + 96];
  float s = v0 + v1 + v2 + v3;
#pragma unroll
  for (int m = 16; m >= 1; m >>= 1) s += __shfl_xor(s, m, 32);
  float mean = s * (1.0f / 128.0f);
  float d0 = v0 - mean, d1 = v1 - mean, d2 = v2 - mean, d3 = v3 - mean;
  float q = d0 * d0 + d1 * d1 + d2 * d2 + d3 * d3;
#pragma unroll
  for (int m = 16; m >= 1; m >>= 1) q += __shfl_xor(q, m, 32);
  float rs = rsqrtf(q * (1.0f / 128.0f) + 1e-5f);
  float o0 = d0 * rs * g[lane]      + b[lane];
  float o1 = d1 * rs * g[lane + 32] + b[lane + 32];
  float o2 = d2 * rs * g[lane + 64] + b[lane + 64];
  float o3 = d3 * rs * g[lane + 96] + b[lane + 96];
  outf[lane]      = o0; outf[lane + 32] = o1;
  outf[lane + 64] = o2; outf[lane + 96] = o3;
  outh[lane]      = (_Float16)o0; outh[lane + 32] = (_Float16)o1;
  outh[lane + 64] = (_Float16)o2; outh[lane + 96] = (_Float16)o3;
}

__global__ __launch_bounds__(256) void ctrl_kernel(GArgs ga) {
  __shared__ float xact[128];                       // residual stream (f32)
  __shared__ float hbuf[128];                       // LN output (f32, head use)
  __shared__ __align__(16) _Float16 hbuf_h[128];    // LN output (f16, GEMM A)
  __shared__ __align__(16) _Float16 attno_h[128];   // attention out (f16)
  __shared__ __align__(16) _Float16 fcb_h[512];     // gelu(fc) (f16)
  __shared__ float qkv_s[384];
  __shared__ float scores_s[4][64];
  __shared__ float Ub[TSEQ + 1];
  __shared__ float xs[8][3];
  __shared__ float ys[8];
  __shared__ float eTok;

  const int tid = threadIdx.x, wv = tid >> 5, lane = tid & 31;

  // ---- Phase 0: swizzle all GEMM weights into f16 WMMA B-fragment layout ---
  for (int mi = 0; mi < 16; ++mi) {
    int L = mi >> 2, m = mi & 3;
    int K = (m == 3) ? 512 : 128;
    int N = (m == 0) ? 384 : (m == 1) ? 128 : (m == 2) ? 512 : 128;
    const float* src = (m == 0) ? ga.wqkv[L] : (m == 1) ? ga.wo[L]
                     : (m == 2) ? ga.wfc[L]  : ga.wpr[L];
    _Float16* dst = ga.frag + (size_t)L * FR_LAYER +
                    ((m == 0) ? FR_QKV_OFF : (m == 1) ? FR_O_OFF
                     : (m == 2) ? FR_FC_OFF : FR_PR_OFF);
    int total = K * N, nk = K >> 5;
    for (int e = tid; e < total; e += 256) {
      int fr = e >> 9, rr = e & 511;
      int l = rr >> 4, j = (rr & 15) >> 1, h = rr & 1;
      int nt = fr / nk, kc = fr - nt * nk;
      int base = (l < 16) ? ((j < 4) ? 2 * j : 2 * j + 8)
                          : ((j < 4) ? 2 * j + 8 : 2 * j + 16);
      int k = kc * 32 + base + h, n = nt * 16 + (l & 15);
      dst[e] = (_Float16)src[k * N + n];
    }
  }
  // ---- init persistent loop state ----
  if (tid <= TSEQ) Ub[tid] = 0.0f;
  if (tid < 8) { ys[tid] = 0.0f; xs[tid][0] = xs[tid][1] = xs[tid][2] = 0.0f; }
  __threadfence_block();
  __syncthreads();

  float* Kc = ga.kv;                 // [L][t][d]
  float* Vc = ga.kv + KV_K_FLOATS;   // [L][t][d]

  // ======================= sequential closed loop ==========================
  for (int i = 0; i < TSEQ; ++i) {
    // record Y[:, i] = y (pre-update); compute e_0 = r[0,i] - y_0
    if (tid < 8) ga.Y[tid * TSEQ + i] = ys[tid];
    if (tid == 0) eTok = ga.r[i] - ys[0];
    __syncthreads();

    // token embedding: concat([e,u]) @ w_in + b_in + wpe[i]
    if (tid < 128)
      xact[tid] = eTok * ga.w_in[tid] + Ub[i] * ga.w_in[128 + tid] +
                  ga.b_in[tid] + ga.wpe[i * 128 + tid];
    __syncthreads();

    for (int L = 0; L < NLAYER; ++L) {
      const _Float16* fL = ga.frag + (size_t)L * FR_LAYER;
      // --- LN1 ---
      ln128(xact, hbuf, hbuf_h, ga.l1g[L], ga.l1b[L]);
      __syncthreads();
      // --- qkv = h @ Wqkv + b (128 -> 384) ---
      mv(fL + FR_QKV_OFF, 4, ga.bqkv[L], hbuf_h, qkv_s, nullptr, 24, MV_STORE);
      __syncthreads();
      // --- append K,V at position i ---
      if (tid < 128) {
        Kc[(size_t)L * 8192 + i * 128 + tid] = qkv_s[128 + tid];
        Vc[(size_t)L * 8192 + i * 128 + tid] = qkv_s[256 + tid];
      }
      __threadfence_block();
      __syncthreads();
      // --- causal attention over t=0..i (waves 0..3 = heads) ---
      if (wv < 4) {
        int hH = wv;
        const float* qh = qkv_s + hH * 32;
        const float* Kb = Kc + (size_t)L * 8192;
        const float* Vb = Vc + (size_t)L * 8192;
        int t0 = lane, t1 = lane + 32;
        float s0 = -1e30f, s1 = -1e30f;
        if (t0 <= i) {
          float s = 0.0f;
#pragma unroll
          for (int d = 0; d < 32; ++d) s += qh[d] * Kb[t0 * 128 + hH * 32 + d];
          s0 = s * 0.17677669529663687f; // 1/sqrt(32)
        }
        if (t1 <= i) {
          float s = 0.0f;
#pragma unroll
          for (int d = 0; d < 32; ++d) s += qh[d] * Kb[t1 * 128 + hH * 32 + d];
          s1 = s * 0.17677669529663687f;
        }
        float mx = fmaxf(s0, s1);
#pragma unroll
        for (int m = 16; m >= 1; m >>= 1) mx = fmaxf(mx, __shfl_xor(mx, m, 32));
        float p0 = (t0 <= i) ? expf(s0 - mx) : 0.0f;
        float p1 = (t1 <= i) ? expf(s1 - mx) : 0.0f;
        float sm = p0 + p1;
#pragma unroll
        for (int m = 16; m >= 1; m >>= 1) sm += __shfl_xor(sm, m, 32);
        float inv = 1.0f / sm;
        scores_s[hH][t0] = p0 * inv;
        scores_s[hH][t1] = p1 * inv;
        // o[d] = sum_t p_t * V[t][d]   (lane = d within head)
        float o = 0.0f;
        for (int t = 0; t <= i; ++t)
          o += scores_s[hH][t] * Vb[t * 128 + hH * 32 + lane];
        attno_h[hH * 32 + lane] = (_Float16)o;
      }
      __syncthreads();
      // --- x += o @ Wo + bo ---
      mv(fL + FR_O_OFF, 4, ga.bo[L], attno_h, xact, nullptr, 8, MV_ADD);
      __syncthreads();
      // --- LN2 + MLP ---
      ln128(xact, hbuf, hbuf_h, ga.l2g[L], ga.l2b[L]);
      __syncthreads();
      mv(fL + FR_FC_OFF, 4, ga.bfc[L], hbuf_h, nullptr, fcb_h, 32, MV_GELU);
      __syncthreads();
      mv(fL + FR_PR_OFF, 16, ga.bpr[L], fcb_h, xact, nullptr, 8, MV_ADD);
      __syncthreads();
    }

    // --- final LN + head (scalar) ---
    ln128(xact, hbuf, hbuf_h, ga.lnfg, ga.lnfb);
    __syncthreads();
    if (wv == 0) {
      float s = hbuf[lane] * ga.whead[lane] + hbuf[lane + 32] * ga.whead[lane + 32] +
                hbuf[lane + 64] * ga.whead[lane + 64] + hbuf[lane + 96] * ga.whead[lane + 96];
#pragma unroll
      for (int m = 16; m >= 1; m >>= 1) s += __shfl_xor(s, m, 32);
      if (lane == 0) Ub[i + 1] = s + ga.bhead[0];
    }
    __syncthreads();

    // --- plant update: x_b = A_b x_b + B_b u_i ; y_b = C_b x_b + D_b u_{i+1} ---
    if (tid < 8) {
      int b = tid;
      const float* Ab = ga.A + b * 9;
      const float* Bb = ga.Bm + b * 3;
      const float* Cb = ga.Cm + b * 3;
      float Db = ga.Dm[b];
      float u0 = Ub[i], u1 = Ub[i + 1];
      float x0 = xs[b][0], x1 = xs[b][1], x2 = xs[b][2];
      float n0 = Ab[0] * x0 + Ab[1] * x1 + Ab[2] * x2 + Bb[0] * u0;
      float n1 = Ab[3] * x0 + Ab[4] * x1 + Ab[5] * x2 + Bb[1] * u0;
      float n2 = Ab[6] * x0 + Ab[7] * x1 + Ab[8] * x2 + Bb[2] * u0;
      xs[b][0] = n0; xs[b][1] = n1; xs[b][2] = n2;
      ys[b] = Cb[0] * n0 + Cb[1] * n1 + Cb[2] * n2 + Db * u1;
    }
    __syncthreads();
  }
}

// ---------------------------------------------------------------------------
// Host: resolve flattened-input order, fill arg struct, launch one workgroup.
// ---------------------------------------------------------------------------
extern "C" void kernel_launch(void* const* d_in, const int* in_sizes, int n_in,
                              void* d_out, int out_size, void* d_ws, size_t ws_size,
                              hipStream_t stream) {
  GArgs ga;
  ga.A  = (const float*)d_in[0];
  ga.Bm = (const float*)d_in[1];
  ga.Cm = (const float*)d_in[2];
  ga.Dm = (const float*)d_in[3];

  // Detect flattening order via in_sizes fingerprint:
  //  INS : A,B,C,D,r(512), w_in(256),b_in,wpe, blocks(insertion), lnf_g,lnf_b,w_head,b_head
  //  MIX : A,B,C,D,r(512), b_head(1),b_in, blocks(sorted), lnf_b,lnf_g,w_head,w_in,wpe
  //  SORT: A,B,C,D, b_head(1),b_in, blocks(sorted), lnf_b,lnf_g,w_head,w_in,wpe, r
  int mode;
  if (in_sizes[4] == 512) mode = (in_sizes[5] == 256) ? 0 : 1;
  else                    mode = 2;

  if (mode == 0) {
    ga.r    = (const float*)d_in[4];
    ga.w_in = (const float*)d_in[5];
    ga.b_in = (const float*)d_in[6];
    ga.wpe  = (const float*)d_in[7];
    for (int L = 0; L < 4; ++L) {
      int base = 8 + 12 * L;
      ga.l1g[L]  = (const float*)d_in[base + 0];
      ga.l1b[L]  = (const float*)d_in[base + 1];
      ga.wqkv[L] = (const float*)d_in[base + 2];
      ga.bqkv[L] = (const float*)d_in[base + 3];
      ga.wo[L]   = (const float*)d_in[base + 4];
      ga.bo[L]   = (const float*)d_in[base + 5];
      ga.l2g[L]  = (const float*)d_in[base + 6];
      ga.l2b[L]  = (const float*)d_in[base + 7];
      ga.wfc[L]  = (const float*)d_in[base + 8];
      ga.bfc[L]  = (const float*)d_in[base + 9];
      ga.wpr[L]  = (const float*)d_in[base + 10];
      ga.bpr[L]  = (const float*)d_in[base + 11];
    }
    ga.lnfg  = (const float*)d_in[56];
    ga.lnfb  = (const float*)d_in[57];
    ga.whead = (const float*)d_in[58];
    ga.bhead = (const float*)d_in[59];
  } else {
    int p = (mode == 1) ? 5 : 4;
    ga.r = (const float*)d_in[(mode == 1) ? 4 : (n_in - 1)];
    ga.bhead = (const float*)d_in[p];
    ga.b_in  = (const float*)d_in[p + 1];
    for (int L = 0; L < 4; ++L) {
      int base = p + 2 + 12 * L;
      ga.bfc[L]  = (const float*)d_in[base + 0];
      ga.bo[L]   = (const float*)d_in[base + 1];
      ga.bpr[L]  = (const float*)d_in[base + 2];
      ga.bqkv[L] = (const float*)d_in[base + 3];
      ga.l1b[L]  = (const float*)d_in[base + 4];
      ga.l1g[L]  = (const float*)d_in[base + 5];
      ga.l2b[L]  = (const float*)d_in[base + 6];
      ga.l2g[L]  = (const float*)d_in[base + 7];
      ga.wfc[L]  = (const float*)d_in[base + 8];
      ga.wo[L]   = (const float*)d_in[base + 9];
      ga.wpr[L]  = (const float*)d_in[base + 10];
      ga.wqkv[L] = (const float*)d_in[base + 11];
    }
    ga.lnfb  = (const float*)d_in[p + 50];
    ga.lnfg  = (const float*)d_in[p + 51];
    ga.whead = (const float*)d_in[p + 52];
    ga.w_in  = (const float*)d_in[p + 53];
    ga.wpe   = (const float*)d_in[p + 54];
  }

  ga.frag = (_Float16*)d_ws;
  ga.kv   = (float*)((char*)d_ws + KV_BYTE_OFF);
  ga.Y    = (float*)d_out;

  ctrl_kernel<<<dim3(1), dim3(256), 0, stream>>>(ga);
}
